// GNN_84774064489089
// MI455X (gfx1250) — matrix operand order repaired and checked by reading, not compiled, not run on hardware
//
#include <hip/hip_runtime.h>

// ---------------------------------------------------------------------------
// DGCNN-style block for MI455X (gfx1250, wave32, WMMA).
// All heavy contractions use v_wmma_f32_16x16x32_f16; fragments are built with
// aligned 16B dwordx4 loads matching the CDNA5 ISA A/B VGPR layouts:
//   A (16x32 f16): lane L -> row M=L&15; e<8 -> K=8*hi+e, e>=8 -> K=16+8*hi+(e-8)
//   B (32x16 f16): lane L -> col N=L&15; same K mapping as A
//   C/D (16x16 f32): lane L, vgpr j -> M=j+8*hi, N=L&15
// GEMM waves are register-blocked 16x(16*NT); NT/MODE are template params so
// the inner loop is a clean  b128-load x4 -> v_wmma xNT  sequence (no selects).
// ---------------------------------------------------------------------------

#define BB 8
#define CC 128
#define NN 2000
#define KK 9
#define C2v 256
#define CIv 32
#define NP (BB * NN)      /* 16000 positions (b,n)         */
#define PE (NP * KK)      /* 144000 edge positions (b,n,k) */
#define PC (NP * 3)       /* 48000 conv1 positions (b,n,j) */

typedef __attribute__((ext_vector_type(16))) _Float16 v16h;
typedef __attribute__((ext_vector_type(8)))  float    v8f;

union FragU { v16h h; uint4 u[2]; };

__device__ __forceinline__ unsigned enc_f32(float f) {
  unsigned b = __float_as_uint(f);
  return (b & 0x80000000u) ? ~b : (b | 0x80000000u);   // order-preserving map
}
__device__ __forceinline__ float dec_f32(unsigned k) {
  unsigned b = (k & 0x80000000u) ? (k & 0x7fffffffu) : ~k;
  return __uint_as_float(b);
}

// ---------------------------------------------------------------------------
// Generic WMMA GEMM: out[M][Nd] = A[M][Kd] * Bt[Nd][Kd]^T + bias
// Each wave computes a 16 x (16*NT) output block: the A fragment for a k-step
// is loaded once and reused by NT consecutive WMMAs.
// MODE 0: plain f32 store.  MODE 1: fused max-over-K(=9 rows) via atomicMax
// on order-preserving-encoded floats into omax[M/9][Nd].
// ---------------------------------------------------------------------------
template <int NT, int MODE>
__global__ void __launch_bounds__(32)
gemm_wmma_f16(const _Float16* __restrict__ A, const _Float16* __restrict__ Bt,
              const float* __restrict__ bias, float* __restrict__ out,
              unsigned* __restrict__ omax, int Nd, int Kd)
{
  const int lane = threadIdx.x;
  const int li = lane & 15;
  const int hi = lane >> 4;
  const long m0 = (long)blockIdx.x * 16;
  const int  n0 = blockIdx.y * 16 * NT;
  const _Float16* arow = A + (m0 + li) * (long)Kd;
  const _Float16* brow[NT];
#pragma unroll
  for (int t = 0; t < NT; ++t)
    brow[t] = Bt + (long)(n0 + 16 * t + li) * Kd;

  v8f acc[NT];
#pragma unroll
  for (int t = 0; t < NT; ++t) acc[t] = (v8f){0.f,0.f,0.f,0.f,0.f,0.f,0.f,0.f};

  for (int k0 = 0; k0 < Kd; k0 += 32) {
    if (k0 + 128 < Kd) __builtin_prefetch(arow + k0 + 128, 0, 3);
    FragU a;
    a.u[0] = *(const uint4*)(arow + k0 + 8 * hi);
    a.u[1] = *(const uint4*)(arow + k0 + 8 * hi + 16);
#pragma unroll
    for (int t = 0; t < NT; ++t) {
      FragU b;
      b.u[0] = *(const uint4*)(brow[t] + k0 + 8 * hi);
      b.u[1] = *(const uint4*)(brow[t] + k0 + 8 * hi + 16);
      acc[t] = __builtin_amdgcn_wmma_f32_16x16x32_f16(false, a.h, false, b.h,
                                                      (short)0, acc[t], false, false);
    }
  }
#pragma unroll
  for (int t = 0; t < NT; ++t) {
    const int col = n0 + 16 * t + li;
    const float bv = bias ? bias[col] : 0.f;
    if (MODE == 0) {
#pragma unroll
      for (int j = 0; j < 8; ++j)
        out[(m0 + j + 8 * hi) * (long)Nd + col] = acc[t][j] + bv;
    } else {
#pragma unroll
      for (int j = 0; j < 8; ++j) {
        long r = m0 + j + 8 * hi;
        atomicMax(&omax[(r / KK) * (long)Nd + col], enc_f32(acc[t][j] + bv));
      }
    }
  }
}

// ---------------------------------------------------------------------------
// Fused KNN: per (b, 16-row block), WMMA Gram tiles -> pd -> register top-9.
// A = xT rows (n-block, K=C=128), Bt = xT rows (m-block).  125 tiles / wave;
// the A fragments are loaded once per workgroup (125x reuse).
// ---------------------------------------------------------------------------
__global__ void __launch_bounds__(32)
knn_topk(const _Float16* __restrict__ xT, const float* __restrict__ xx,
         int* __restrict__ idx)
{
  __shared__ float tile[256];
  const int lane = threadIdx.x;
  const int li = lane & 15, hi = lane >> 4;
  const int b  = blockIdx.x / (NN / 16);
  const int n0 = (blockIdx.x % (NN / 16)) * 16;
  const _Float16* arow = xT + ((long)b * NN + n0 + li) * CC;
  FragU a[4];
#pragma unroll
  for (int s = 0; s < 4; ++s) {
    a[s].u[0] = *(const uint4*)(arow + s * 32 + 8 * hi);
    a[s].u[1] = *(const uint4*)(arow + s * 32 + 8 * hi + 16);
  }
  float xn[8];
#pragma unroll
  for (int j = 0; j < 8; ++j) xn[j] = xx[b * NN + n0 + j + 8 * hi];

  float tv[KK]; int ti[KK];
#pragma unroll
  for (int q = 0; q < KK; ++q) { tv[q] = -3.4e38f; ti[q] = 0; }

  for (int mt = 0; mt < NN / 16; ++mt) {
    const int m0 = mt * 16;
    const _Float16* brow = xT + ((long)b * NN + m0 + li) * CC;
    v8f acc = {0.f, 0.f, 0.f, 0.f, 0.f, 0.f, 0.f, 0.f};
#pragma unroll
    for (int s = 0; s < 4; ++s) {
      FragU bf;
      bf.u[0] = *(const uint4*)(brow + s * 32 + 8 * hi);
      bf.u[1] = *(const uint4*)(brow + s * 32 + 8 * hi + 16);
      acc = __builtin_amdgcn_wmma_f32_16x16x32_f16(false, a[s].h, false, bf.h,
                                                   (short)0, acc, false, false);
    }
    const float xm = xx[b * NN + m0 + li];
#pragma unroll
    for (int j = 0; j < 8; ++j)
      tile[(j + 8 * hi) * 16 + li] = 2.f * acc[j] - xn[j] - xm;
    __syncthreads();
    if (lane < 16) {
      for (int mm = 0; mm < 16; ++mm) {
        float cv = tile[lane * 16 + mm];
        if (cv > tv[KK - 1]) {                 // branchless bubble insert (regs)
          int ci = m0 + mm;
#pragma unroll
          for (int q = 0; q < KK; ++q) {
            if (cv > tv[q]) {
              float ft = tv[q]; tv[q] = cv; cv = ft;
              int   it = ti[q]; ti[q] = ci; ci = it;
            }
          }
        }
      }
    }
    __syncthreads();
  }
  if (lane < 16) {
    long base = ((long)b * NN + n0 + lane) * KK;
#pragma unroll
    for (int q = 0; q < KK; ++q) idx[base + q] = ti[q];
  }
}

// ---------------------------------------------------------------------------
// Prep: xT[b][n][c] f16 (GEMM operand layout) + xx[b][n] = sum_c x^2.
// ---------------------------------------------------------------------------
__global__ void prep_xt(const float* __restrict__ F, _Float16* __restrict__ xT,
                        float* __restrict__ xx)
{
  int t = blockIdx.x * blockDim.x + threadIdx.x;
  if (t >= NP) return;
  int b = t / NN, n = t % NN;
  const float* src = F + (long)b * CC * NN + n;
  float s = 0.f;
  for (int c = 0; c < CC; ++c) {
    float v = src[(long)c * NN];
    s += v * v;
    xT[(long)t * CC + c] = (_Float16)v;
  }
  xx[t] = s;
}

// edge[b][n][k][c2] f16, c2 = [center | center-neighbor]
__global__ void edge_build(const float* __restrict__ F, const int* __restrict__ idx,
                           _Float16* __restrict__ edge)
{
  long pos = blockIdx.x;
  int  c   = threadIdx.x;           // 128 threads
  int  b   = (int)(pos / (NN * KK));
  int  r   = (int)(pos % (NN * KK));
  int  n   = r / KK;
  int  j   = idx[pos];
  const float* Fb = F + (long)(b * CC + c) * NN;
  float ctr = Fb[n];
  float nb  = Fb[j];
  edge[pos * C2v + c]       = (_Float16)ctr;
  edge[pos * C2v + CC + c]  = (_Float16)(ctr - nb);
}

// ---------------------------------------------------------------------------
// Per-(b,c) biased mean/var over S samples of position-major [Bn*S][Cd] data.
// Per-channel BN stats = same kernel with Bn=1.
// ---------------------------------------------------------------------------
template <typename T>
__global__ void __launch_bounds__(256)
stats_bc(const T* __restrict__ src, int S, int Cd,
         float* __restrict__ mean, float* __restrict__ var)
{
  const int blk = blockIdx.x;
  const int c = blk % Cd, b = blk / Cd;
  float s = 0.f, s2 = 0.f;
  for (int i = threadIdx.x; i < S; i += 256) {
    float v = (float)src[((long)b * S + i) * Cd + c];
    s += v; s2 += v * v;
  }
  __shared__ float rs[256], rq[256];
  rs[threadIdx.x] = s; rq[threadIdx.x] = s2;
  __syncthreads();
  for (int st = 128; st > 0; st >>= 1) {
    if (threadIdx.x < st) { rs[threadIdx.x] += rs[threadIdx.x + st];
                            rq[threadIdx.x] += rq[threadIdx.x + st]; }
    __syncthreads();
  }
  if (threadIdx.x == 0) {
    float m = rs[0] / (float)S;
    mean[blk] = m;
    var[blk]  = rq[0] / (float)S - m * m;
  }
}

// _mlps head: InstanceNorm(eps 1e-3) -> BN(eps 1e-5) folded analytically:
// BN mean of inorm output is exactly 0; BN var = avg_b v/(v+1e-3).
__global__ void fin_inorm_bn(const float* __restrict__ mean, const float* __restrict__ var,
                             const float* __restrict__ g, const float* __restrict__ be,
                             int Bn, int Cd, float* __restrict__ alpha, float* __restrict__ beta)
{
  int c = blockIdx.x * blockDim.x + threadIdx.x;
  if (c >= Cd) return;
  float vb = 0.f;
  for (int b = 0; b < Bn; ++b) { float v = var[b * Cd + c]; vb += v / (v + 1e-3f); }
  vb /= (float)Bn;
  float s2 = rsqrtf(vb + 1e-5f);
  for (int b = 0; b < Bn; ++b) {
    float a = g[c] * s2 * rsqrtf(var[b * Cd + c] + 1e-3f);
    alpha[b * Cd + c] = a;
    beta [b * Cd + c] = be[c] - a * mean[b * Cd + c];
  }
}

// Plain training-mode BN fold: alpha = g/sqrt(v+1e-5), beta = be - alpha*m.
__global__ void fin_bn(const float* __restrict__ mean, const float* __restrict__ var,
                       const float* __restrict__ g, const float* __restrict__ be,
                       int Cd, float* __restrict__ alpha, float* __restrict__ beta)
{
  int c = blockIdx.x * blockDim.x + threadIdx.x;
  if (c >= Cd) return;
  float a = g[c] * rsqrtf(var[c] + 1e-5f);
  alpha[c] = a;
  beta[c]  = be[c] - a * mean[c];
}

// y = [relu](alpha[b][c]*x + beta[b][c]) over position-major data.
template <typename TS, typename TD>
__global__ void transform_affine(const TS* __restrict__ src, TD* __restrict__ dst,
                                 const float* __restrict__ alpha, const float* __restrict__ beta,
                                 long total, int S, int Cd, int relu)
{
  long t = (long)blockIdx.x * blockDim.x + threadIdx.x;
  if (t >= total) return;
  int  c = (int)(t % Cd);
  long p = t / Cd;
  int  b = (int)(p / S);
  float v = alpha[b * Cd + c] * (float)src[t] + beta[b * Cd + c];
  if (relu) v = fmaxf(v, 0.f);
  dst[t] = (TD)v;
}

__global__ void decode_max(const unsigned* __restrict__ src, float* __restrict__ dst, long total)
{
  long t = (long)blockIdx.x * blockDim.x + threadIdx.x;
  if (t < total) dst[t] = dec_f32(src[t]);
}

__global__ void combine_xa(const float* __restrict__ xfm, const float* __restrict__ xfa,
                           float* __restrict__ xaf, _Float16* __restrict__ xah, long total)
{
  long t = (long)blockIdx.x * blockDim.x + threadIdx.x;
  if (t >= total) return;
  float v = xfm[t] + xfa[t];
  xaf[t] = v;
  xah[t] = (_Float16)v;
}

__global__ void w_copy_f16(const float* __restrict__ src, _Float16* __restrict__ dst, int n)
{
  int t = blockIdx.x * blockDim.x + threadIdx.x;
  if (t < n) dst[t] = (_Float16)src[t];
}

// W[o][c][t] (o,c in 0..255, t in 0..2) -> Bt[o][t*256+c] f16 for K=768 GEMM.
__global__ void w_conv_reorder(const float* __restrict__ W, _Float16* __restrict__ dst)
{
  int t = blockIdx.x * blockDim.x + threadIdx.x;
  if (t >= C2v * 768) return;
  int o = t / 768, q = t % 768;
  int tk = q / C2v, c = q % C2v;
  dst[t] = (_Float16)W[(long)o * 768 + c * 3 + tk];
}

// AFF global path: ga[8][128] -> 1x1 -> BN(batch of 8) -> relu -> 1x1 -> BN.
__global__ void __launch_bounds__(256)
aff_global_k(const float* __restrict__ ga,
             const float* __restrict__ Wg1, const float* __restrict__ bg1,
             const float* __restrict__ gg1, const float* __restrict__ beg1,
             const float* __restrict__ Wg2, const float* __restrict__ bg2,
             const float* __restrict__ gg2, const float* __restrict__ beg2,
             float* __restrict__ xg)
{
  __shared__ float h1[BB * CIv], h2s[BB * CC], al[CC], bl[CC];
  int t = threadIdx.x;
  { // h1 = Wg1 * ga + bg1            (8*32 outputs, 256 threads)
    int b = t / CIv, ci = t % CIv;
    float s = bg1[ci];
    for (int c = 0; c < CC; ++c) s += Wg1[ci * CC + c] * ga[b * CC + c];
    h1[t] = s;
  }
  __syncthreads();
  if (t < CIv) { // BN over batch of 8
    float m = 0.f;
    for (int b = 0; b < BB; ++b) m += h1[b * CIv + t];
    m /= (float)BB;
    float v = 0.f;
    for (int b = 0; b < BB; ++b) { float d = h1[b * CIv + t] - m; v += d * d; }
    v /= (float)BB;
    float a = gg1[t] * rsqrtf(v + 1e-5f);
    al[t] = a; bl[t] = beg1[t] - a * m;
  }
  __syncthreads();
  { int ci = t % CIv; h1[t] = fmaxf(al[ci] * h1[t] + bl[ci], 0.f); }
  __syncthreads();
  for (int o = t; o < BB * CC; o += 256) { // h2 = Wg2 * relu(...) + bg2
    int b = o / CC, co = o % CC;
    float s = bg2[co];
    for (int ci = 0; ci < CIv; ++ci) s += Wg2[co * CIv + ci] * h1[b * CIv + ci];
    h2s[o] = s;
  }
  __syncthreads();
  if (t < CC) { // BN over batch of 8
    float m = 0.f;
    for (int b = 0; b < BB; ++b) m += h2s[b * CC + t];
    m /= (float)BB;
    float v = 0.f;
    for (int b = 0; b < BB; ++b) { float d = h2s[b * CC + t] - m; v += d * d; }
    v /= (float)BB;
    float a = gg2[t] * rsqrtf(v + 1e-5f);
    al[t] = a; bl[t] = beg2[t] - a * m;
  }
  __syncthreads();
  for (int o = t; o < BB * CC; o += 256) { int co = o % CC; xg[o] = al[co] * h2s[o] + bl[co]; }
}

// Final AFF gate + transpose to [B][C][N][1] output layout.
__global__ void aff_final(const float* __restrict__ hl2,
                          const float* __restrict__ a_l2, const float* __restrict__ b_l2,
                          const float* __restrict__ xg,
                          const float* __restrict__ xfm, const float* __restrict__ xfa,
                          float* __restrict__ out)
{
  long t = (long)blockIdx.x * blockDim.x + threadIdx.x;
  if (t >= (long)BB * CC * NN) return;
  int b = (int)(t / (CC * NN));
  int r = (int)(t % (CC * NN));
  int c = r / NN, n = r % NN;
  long p = (long)b * NN + n;
  float xl = a_l2[c] * hl2[p * CC + c] + b_l2[c];
  float w  = 1.f / (1.f + __expf(-(xl + xg[b * CC + c])));
  out[t] = 2.f * xfm[p * CC + c] * w + 2.f * xfa[p * CC + c] * (1.f - w);
}

// ---------------------------------------------------------------------------
extern "C" void kernel_launch(void* const* d_in, const int* in_sizes, int n_in,
                              void* d_out, int out_size, void* d_ws, size_t ws_size,
                              hipStream_t stream)
{
  (void)in_sizes; (void)n_in; (void)out_size; (void)ws_size;
  // ---- inputs (setup_inputs() dict order, recursively flattened) ----
  const float* F     = (const float*)d_in[0];
  const float* cW1   = (const float*)d_in[1];
  const float* cb1   = (const float*)d_in[2];
  const float* cg1   = (const float*)d_in[3];
  const float* cbe1  = (const float*)d_in[4];
  const float* cW2   = (const float*)d_in[5];
  const float* cb2   = (const float*)d_in[6];
  const float* cg2   = (const float*)d_in[7];
  const float* cbe2  = (const float*)d_in[8];
  const float* m1g   = (const float*)d_in[9];
  const float* m1b   = (const float*)d_in[10];
  const float* m1W   = (const float*)d_in[11];
  const float* m1bi  = (const float*)d_in[12];
  const float* c1g   = (const float*)d_in[13];
  const float* c1b   = (const float*)d_in[14];
  const float* c1W   = (const float*)d_in[15];
  const float* c1bi  = (const float*)d_in[16];
  const float* c2g   = (const float*)d_in[17];
  const float* c2b   = (const float*)d_in[18];
  const float* c2W   = (const float*)d_in[19];
  const float* c2bi  = (const float*)d_in[20];
  const float* Wl1   = (const float*)d_in[21];
  const float* bl1   = (const float*)d_in[22];
  const float* gl1   = (const float*)d_in[23];
  const float* bel1  = (const float*)d_in[24];
  const float* Wl2   = (const float*)d_in[25];
  const float* bl2   = (const float*)d_in[26];
  const float* gl2   = (const float*)d_in[27];
  const float* bel2  = (const float*)d_in[28];
  const float* Wg1   = (const float*)d_in[29];
  const float* bg1   = (const float*)d_in[30];
  const float* gg1   = (const float*)d_in[31];
  const float* beg1  = (const float*)d_in[32];
  const float* Wg2   = (const float*)d_in[33];
  const float* bg2   = (const float*)d_in[34];
  const float* gg2   = (const float*)d_in[35];
  const float* beg2  = (const float*)d_in[36];

  // ---- workspace carve-out (256B aligned) ----
  char* wsb = (char*)d_ws;
  size_t cur = 0;
  auto alloc = [&](size_t bytes) -> void* {
    cur = (cur + 255) & ~(size_t)255;
    void* p = wsb + cur; cur += bytes; return p;
  };
  _Float16* xT    = (_Float16*)alloc((size_t)NP * CC * 2);
  float*    xx    = (float*)   alloc((size_t)NP * 4);
  int*      idx   = (int*)     alloc((size_t)PE * 4);
  _Float16* edge  = (_Float16*)alloc((size_t)PE * C2v * 2);
  _Float16* eact  = (_Float16*)alloc((size_t)PE * C2v * 2);
  _Float16* w1t   = (_Float16*)alloc((size_t)C2v * 768 * 2);
  _Float16* w2t   = (_Float16*)alloc((size_t)C2v * 768 * 2);
  _Float16* wm1   = (_Float16*)alloc((size_t)C2v * C2v * 2);
  _Float16* wc1   = (_Float16*)alloc((size_t)CC * C2v * 2);
  _Float16* wc2   = (_Float16*)alloc((size_t)CC * C2v * 2);
  _Float16* wl1h  = (_Float16*)alloc((size_t)CIv * CC * 2);
  _Float16* wl2h  = (_Float16*)alloc((size_t)CC * CIv * 2);
  float*    h1    = (float*)   alloc((size_t)PC * C2v * 4);
  _Float16* act1  = (_Float16*)alloc((size_t)PC * C2v * 2);
  float*    h2    = (float*)   alloc((size_t)NP * C2v * 4);
  float*    act2f = (float*)   alloc((size_t)NP * C2v * 4);
  _Float16* act2h = (_Float16*)alloc((size_t)NP * C2v * 2);
  unsigned* menc  = (unsigned*)alloc((size_t)NP * C2v * 4);
  float*    omf   = (float*)   alloc((size_t)NP * C2v * 4);
  _Float16* omh   = (_Float16*)alloc((size_t)NP * C2v * 2);
  float*    xfa   = (float*)   alloc((size_t)NP * CC * 4);
  float*    xfm   = (float*)   alloc((size_t)NP * CC * 4);
  float*    xaf   = (float*)   alloc((size_t)NP * CC * 4);
  _Float16* xah   = (_Float16*)alloc((size_t)NP * CC * 2);
  float*    hl1   = (float*)   alloc((size_t)NP * CIv * 4);
  _Float16* al1h  = (_Float16*)alloc((size_t)NP * CIv * 2);
  float*    hl2   = (float*)   alloc((size_t)NP * CC * 4);
  // small stats / folded-affine buffers
  float* st_m  = (float*)alloc(BB * C2v * 4);
  float* st_v  = (float*)alloc(BB * C2v * 4);
  float* af_a  = (float*)alloc(BB * C2v * 4);
  float* af_b  = (float*)alloc(BB * C2v * 4);
  float* ga_m  = (float*)alloc(BB * CC * 4);
  float* ga_v  = (float*)alloc(BB * CC * 4);
  float* xg    = (float*)alloc(BB * CC * 4);
  float* a_l2  = (float*)alloc(CC * 4);
  float* b_l2  = (float*)alloc(CC * 4);
  float* out_f = (float*)d_out;

  auto g1d = [](long n) { return (unsigned)((n + 255) / 256); };

  // ---- weight prep (f16, Bt layouts) ----
  w_conv_reorder<<<g1d(C2v * 768), 256, 0, stream>>>(cW1, w1t);
  w_conv_reorder<<<g1d(C2v * 768), 256, 0, stream>>>(cW2, w2t);
  w_copy_f16<<<g1d(C2v * C2v), 256, 0, stream>>>(m1W, wm1, C2v * C2v);
  w_copy_f16<<<g1d(CC * C2v), 256, 0, stream>>>(c1W, wc1, CC * C2v);
  w_copy_f16<<<g1d(CC * C2v), 256, 0, stream>>>(c2W, wc2, CC * C2v);
  w_copy_f16<<<g1d(CIv * CC), 256, 0, stream>>>(Wl1, wl1h, CIv * CC);
  w_copy_f16<<<g1d(CC * CIv), 256, 0, stream>>>(Wl2, wl2h, CC * CIv);

  // ---- KNN + edge features ----
  prep_xt<<<g1d(NP), 256, 0, stream>>>(F, xT, xx);
  knn_topk<<<BB * (NN / 16), 32, 0, stream>>>(xT, xx, idx);
  edge_build<<<PE, CC, 0, stream>>>(F, idx, edge);

  // ---- mlp1: inorm+BN(analytic)+relu -> 1x1 conv, fused max-over-k ----
  stats_bc<_Float16><<<BB * C2v, 256, 0, stream>>>(edge, NN * KK, C2v, st_m, st_v);
  fin_inorm_bn<<<1, 256, 0, stream>>>(st_m, st_v, m1g, m1b, BB, C2v, af_a, af_b);
  transform_affine<_Float16, _Float16><<<g1d((long)PE * C2v), 256, 0, stream>>>(
      edge, eact, af_a, af_b, (long)PE * C2v, NN * KK, C2v, 1);
  hipMemsetAsync(menc, 0, (size_t)NP * C2v * 4, stream);
  gemm_wmma_f16<4, 1><<<dim3(PE / 16, C2v / 64), 32, 0, stream>>>(
      eact, wm1, m1bi, nullptr, menc, C2v, C2v);

  // ---- conv branch: conv1(K=768) -> BN -> relu -> conv2(K=768) -> BN -> relu
  gemm_wmma_f16<4, 0><<<dim3(PC / 16, C2v / 64), 32, 0, stream>>>(
      edge, w1t, cb1, h1, nullptr, C2v, 768);
  stats_bc<float><<<C2v, 256, 0, stream>>>(h1, PC, C2v, st_m, st_v);
  fin_bn<<<1, 256, 0, stream>>>(st_m, st_v, cg1, cbe1, C2v, af_a, af_b);
  transform_affine<float, _Float16><<<g1d((long)PC * C2v), 256, 0, stream>>>(
      h1, act1, af_a, af_b, (long)PC * C2v, PC, C2v, 1);
  gemm_wmma_f16<4, 0><<<dim3(NP / 16, C2v / 64), 32, 0, stream>>>(
      act1, w2t, cb2, h2, nullptr, C2v, 768);
  stats_bc<float><<<C2v, 256, 0, stream>>>(h2, NP, C2v, st_m, st_v);
  fin_bn<<<1, 256, 0, stream>>>(st_m, st_v, cg2, cbe2, C2v, af_a, af_b);
  transform_affine<float, float><<<g1d((long)NP * C2v), 256, 0, stream>>>(
      h2, act2f, af_a, af_b, (long)NP * C2v, NP, C2v, 1);

  // ---- change1 on conv-branch output -> residual xfa [NP][128] ----
  stats_bc<float><<<BB * C2v, 256, 0, stream>>>(act2f, NN, C2v, st_m, st_v);
  fin_inorm_bn<<<1, 256, 0, stream>>>(st_m, st_v, c1g, c1b, BB, C2v, af_a, af_b);
  transform_affine<float, _Float16><<<g1d((long)NP * C2v), 256, 0, stream>>>(
      act2f, act2h, af_a, af_b, (long)NP * C2v, NN, C2v, 1);
  gemm_wmma_f16<4, 0><<<dim3(NP / 16, CC / 64), 32, 0, stream>>>(
      act2h, wc1, c1bi, xfa, nullptr, CC, C2v);

  // ---- change2 on max-pooled mlp1 output -> xfm [NP][128] ----
  decode_max<<<g1d((long)NP * C2v), 256, 0, stream>>>(menc, omf, (long)NP * C2v);
  stats_bc<float><<<BB * C2v, 256, 0, stream>>>(omf, NN, C2v, st_m, st_v);
  fin_inorm_bn<<<1, 256, 0, stream>>>(st_m, st_v, c2g, c2b, BB, C2v, af_a, af_b);
  transform_affine<float, _Float16><<<g1d((long)NP * C2v), 256, 0, stream>>>(
      omf, omh, af_a, af_b, (long)NP * C2v, NN, C2v, 1);
  gemm_wmma_f16<4, 0><<<dim3(NP / 16, CC / 64), 32, 0, stream>>>(
      omh, wc2, c2bi, xfm, nullptr, CC, C2v);

  // ---- AFF: xa = x + residual; local attention path (WMMA) ----
  combine_xa<<<g1d((long)NP * CC), 256, 0, stream>>>(xfm, xfa, xaf, xah, (long)NP * CC);
  stats_bc<float><<<BB * CC, 256, 0, stream>>>(xaf, NN, CC, ga_m, ga_v);  // ga = mean
  gemm_wmma_f16<2, 0><<<dim3(NP / 16, 1), 32, 0, stream>>>(
      xah, wl1h, bl1, hl1, nullptr, CIv, CC);
  stats_bc<float><<<CIv, 256, 0, stream>>>(hl1, NP, CIv, st_m, st_v);
  fin_bn<<<1, 256, 0, stream>>>(st_m, st_v, gl1, bel1, CIv, af_a, af_b);
  transform_affine<float, _Float16><<<g1d((long)NP * CIv), 256, 0, stream>>>(
      hl1, al1h, af_a, af_b, (long)NP * CIv, NP, CIv, 1);
  gemm_wmma_f16<4, 0><<<dim3(NP / 16, CC / 64), 32, 0, stream>>>(
      al1h, wl2h, bl2, hl2, nullptr, CC, CIv);
  stats_bc<float><<<CC, 256, 0, stream>>>(hl2, NP, CC, st_m, st_v);
  fin_bn<<<1, 256, 0, stream>>>(st_m, st_v, gl2, bel2, CC, a_l2, b_l2);

  // ---- AFF global path + final gate ----
  aff_global_k<<<1, 256, 0, stream>>>(ga_m, Wg1, bg1, gg1, beg1, Wg2, bg2, gg2, beg2, xg);
  aff_final<<<g1d((long)BB * CC * NN), 256, 0, stream>>>(
      hl2, a_l2, b_l2, xg, xfm, xfa, out_f);
}
